// AutocorrSeq_14843406975599
// MI455X (gfx1250) — matrix operand order repaired and checked by reading, not compiled, not run on hardware
//
#include <hip/hip_runtime.h>

// CDNA5 / gfx1250: wave32, WMMA f32 = f16 x f16 16x16x32.
typedef __attribute__((ext_vector_type(16))) _Float16 v16h;
typedef __attribute__((ext_vector_type(4)))  _Float16 v4h;
typedef __attribute__((ext_vector_type(8)))  float    v8f;

#define AC_T     8192            // sequence length (from reference)
#define NLAG     64              // lags 1..64 output, lag 0 for normalization
#define PADL     16              // zero pad left  (A needs xc[s-r], r<=15)
#define PADR     96              // zero pad right (B reads xc[s+16q], up to 64 past the row)
#define NTHREADS 256             // 8 waves of 32
#define WAVESZ   32
#define NWAVES   (NTHREADS / WAVESZ)
#define KCHUNK   32              // K per WMMA (f16 path)
#define NCHUNKS  (AC_T / KCHUNK) // 256
#define CPW      (NCHUNKS / NWAVES) // 32 contiguous chunks per wave

// D[r,q] = sum_s xc[s-r] * xc[s+16q]   ->   ac[16q + r]
__launch_bounds__(NTHREADS)
__global__ void autocorr64_wmma(const float* __restrict__ X,
                                float* __restrict__ out) {
  // xf: f32 staging for the row (single global read). Dead after pass 2,
  // so its first 8KB is reused as the per-wave D-tile scratch.
  __shared__ float    xf[AC_T];                   // 32 KB
  __shared__ _Float16 xc[PADL + AC_T + PADR];     // centered row, f16 (~16.6 KB)
  __shared__ float    wsum[NWAVES];

  float* dtile = xf;                              // [NWAVES][256], aliases xf

  const int b    = blockIdx.x;
  const int tid  = threadIdx.x;
  const int lane = tid & (WAVESZ - 1);
  const int wv   = tid / WAVESZ;
  const float* __restrict__ row = X + (size_t)b * AC_T;

  // ---- pass 1: stage row into LDS (float4) + f32 mean ----
  float s = 0.f;
  for (int i = tid; i < AC_T / 4; i += NTHREADS) {
    float4 v = reinterpret_cast<const float4*>(row)[i];
    reinterpret_cast<float4*>(xf)[i] = v;
    s += (v.x + v.y) + (v.z + v.w);
  }
#pragma unroll
  for (int off = 16; off > 0; off >>= 1) s += __shfl_xor(s, off, WAVESZ);
  if (lane == 0) wsum[wv] = s;
  __syncthreads();
  float mean = 0.f;
#pragma unroll
  for (int w = 0; w < NWAVES; ++w) mean += wsum[w];
  mean *= (1.0f / AC_T);

  // ---- pass 2: center + f16 convert from LDS, zero pads ----
  for (int i = tid; i < PADL; i += NTHREADS) xc[i] = (_Float16)0.f;
  for (int i = tid; i < PADR; i += NTHREADS) xc[PADL + AC_T + i] = (_Float16)0.f;
  for (int i = tid; i < AC_T / 4; i += NTHREADS) {
    float4 v = reinterpret_cast<float4*>(xf)[i];
    v4h h;
    h[0] = (_Float16)(v.x - mean);
    h[1] = (_Float16)(v.y - mean);
    h[2] = (_Float16)(v.z - mean);
    h[3] = (_Float16)(v.w - mean);
    reinterpret_cast<v4h*>(xc + PADL)[i] = h;    // 8-byte aligned DS stores
  }
  __syncthreads();   // also guarantees xf reads are done before dtile reuse

  // ---- WMMA accumulation ----
  // A (16x32 f16): lane L holds row m = L%16; VGPR v packs K pair
  //   K = (v/4)*16 + (L/16)*8 + (v%4)*2 + {0,1}   (ISA 7.12.2, 16-bit A layout)
  // B (32x16 f16): lane L holds column n = L%16; VGPR v packs the same K pair as A.
  // B(k, n) = xc[s0 + k + 16n], reloaded from LDS every chunk in lanes with n < 5:
  // 16 LDS loads per lane per chunk; columns 5..15 stay pinned at zero.
  const int m     = lane & 15;
  const int half  = lane >> 4;
  const int sbase = wv * CPW * KCHUNK;           // contiguous chunk range per wave

  v16h bfrag = {};                               // upper columns stay zero

  v8f acc0 = {}, acc1 = {};
#pragma unroll 4
  for (int i = 0; i < CPW; ++i) {
    const int s0 = sbase + i * KCHUNK;

    v16h afrag;
#pragma unroll
    for (int v = 0; v < 8; ++v) {
      const int k   = (v >> 2) * 16 + half * 8 + (v & 3) * 2;
      const int idx = PADL + s0 + k - m;         // xc[s - r], r = m
      afrag[2 * v]     = xc[idx];
      afrag[2 * v + 1] = xc[idx + 1];
      if (m < 5) {
        const int bidx = PADL + s0 + k + 16 * m;
        bfrag[2 * v]     = xc[bidx];
        bfrag[2 * v + 1] = xc[bidx + 1];
      }
    }

    // (neg_a, A, neg_b, B, c_mod, C, reuse_a, reuse_b)
    if (i & 1)
      acc1 = __builtin_amdgcn_wmma_f32_16x16x32_f16(false, afrag, false, bfrag,
                                                    (short)0, acc1, false, false);
    else
      acc0 = __builtin_amdgcn_wmma_f32_16x16x32_f16(false, afrag, false, bfrag,
                                                    (short)0, acc0, false, false);


  }

  // Spill per-wave D tile: D[m,n] lives in vgpr j = m%8, lane = n + 16*(m/8)
#pragma unroll
  for (int j = 0; j < 8; ++j) {
    const int mrow = j + 8 * half;
    dtile[wv * 256 + mrow * 16 + m] = acc0[j] + acc1[j];
  }
  __syncthreads();

  // ---- cross-wave sum + normalize by lag-0 (biased denom cancels) ----
  if (tid >= 1 && tid <= NLAG) {
    const int k = tid;
    const int r = k & 15, q = k >> 4;
    float ack = 0.f, ac0 = 0.f;
#pragma unroll
    for (int w = 0; w < NWAVES; ++w) {
      ack += dtile[w * 256 + r * 16 + q];
      ac0 += dtile[w * 256];
    }
    out[(size_t)b * NLAG + (k - 1)] = ack / ac0;
  }
}

extern "C" void kernel_launch(void* const* d_in, const int* in_sizes, int n_in,
                              void* d_out, int out_size, void* d_ws, size_t ws_size,
                              hipStream_t stream) {
  (void)n_in; (void)d_ws; (void)ws_size; (void)out_size;
  const float* X   = (const float*)d_in[0];
  float*       out = (float*)d_out;
  const int B = in_sizes[0] / AC_T;     // 4096 rows
  autocorr64_wmma<<<B, NTHREADS, 0, stream>>>(X, out);
}
